// Net_17617955848347
// MI455X (gfx1250) — matrix operand order repaired and checked by reading, not compiled, not run on hardware
//
#include <hip/hip_runtime.h>
#include <hip/hip_bf16.h>
#include <math.h>

// Problem constants (match reference)
#define NN   50000
#define NE   800000
#define INF_ 128
#define HIDD 64
#define NH   4
#define HDIM 256   // NH*HIDD

// ---- WMMA fragment types (gfx1250, wave32) ----
typedef __attribute__((ext_vector_type(16))) __bf16         v16bf;
typedef __attribute__((ext_vector_type(8)))  float          v8f;
typedef __attribute__((ext_vector_type(16))) unsigned short u16x16;
typedef __attribute__((ext_vector_type(8)))  unsigned short u16x8;

// fp32 -> bf16 (round to nearest even), stored as u16
__device__ __forceinline__ unsigned short f2bf(float f) {
  unsigned u = __float_as_uint(f);
  return (unsigned short)((u + 0x7FFFu + ((u >> 16) & 1u)) >> 16);
}

// order-preserving float <-> uint for atomicMax-based segment max
__device__ __forceinline__ unsigned fenc(float f) {
  unsigned u = __float_as_uint(f);
  return (u & 0x80000000u) ? ~u : (u | 0x80000000u);
}
__device__ __forceinline__ float fdec(unsigned k) {
  return __uint_as_float((k & 0x80000000u) ? (k & 0x7FFFFFFFu) : ~k);
}

// ---------------- elementwise helpers ----------------
__global__ void k_cvt_bf16(const float* __restrict__ s, unsigned short* __restrict__ d, int n) {
  int i = blockIdx.x * blockDim.x + threadIdx.x;
  if (i < n) d[i] = f2bf(s[i]);
}

__global__ void k_relu_cvt(float* __restrict__ x, unsigned short* __restrict__ xb, int n) {
  int i = blockIdx.x * blockDim.x + threadIdx.x;
  if (i < n) { float v = x[i]; v = v > 0.f ? v : 0.f; x[i] = v; xb[i] = f2bf(v); }
}

// ---------------- weight packing into WMMA B-fragment layout ----------------
// B fragment (32x16 bf16, 16x16x32 WMMA): lanes 0-15 hold K=0..7,16..23 of column n=lane,
// lanes 16-31 hold K=8..15,24..31 of column n=lane-16; element i<8 = K kb+i, i>=8 = K kb+8+i.
// Packed layout: P[((kt*(Ncols/16) + ct)*32 + lane)*16 + i]  (32B-contiguous per lane).
__global__ void k_pack_w(const float* __restrict__ W, unsigned short* __restrict__ P,
                         int K, int Ncols) {
  int idx = blockIdx.x * blockDim.x + threadIdx.x;   // one thread per (kt, ct, lane)
  int nct = Ncols >> 4;
  int total = (K >> 5) * nct * 32;
  if (idx >= total) return;
  int lane = idx & 31;
  int ct   = (idx >> 5) % nct;
  int kt   = idx / (32 * nct);
  int kb   = (lane >= 16) ? 8 : 0;
  int n    = lane & 15;
  const float* wp = W + (size_t)(kt * 32 + kb) * Ncols + ct * 16 + n;
  unsigned short* op = P + (size_t)idx * 16;
#pragma unroll
  for (int i = 0; i < 16; i++) op[i] = f2bf(wp[(size_t)(i < 8 ? i : i + 8) * Ncols]);
}

// ---------------- WMMA GEMM: C[M,256] = A[M,K](bf16) x B[K,256](packed bf16) ----------------
// block = 128 threads (4 waves); block covers 32 rows x 256 cols; wave covers 32x64.
// Each B fragment (2x global_load_b128) feeds two WMMAs (rows 0-15 and 16-31).
// Out-of-range rows in the last block are CLAMPED on load (junk computed, never
// stored) so the k-loop contains no exec-mask branches.
__global__ void k_gemm_bf16_n256(const unsigned short* __restrict__ A,
                                 const unsigned short* __restrict__ BP,
                                 float* __restrict__ C, int K) {
  const int lane  = threadIdx.x & 31;
  const int wave  = threadIdx.x >> 5;
  const int row0  = blockIdx.x * 32;
  const int colt0 = wave * 4;             // first 16-col tile of this wave
  const int n     = lane & 15;
  const int ka    = (lane >= 16) ? 8 : 0; // A K-half offset
  const int nkt   = K >> 5;

  int rA0 = row0 + n;
  int rA1 = row0 + 16 + n;
  rA0 = rA0 < NN ? rA0 : NN - 1;          // clamp: junk rows computed, not stored
  rA1 = rA1 < NN ? rA1 : NN - 1;
  const unsigned short* arow0 = A + (size_t)rA0 * K;
  const unsigned short* arow1 = A + (size_t)rA1 * K;

  v8f acc[2][4] = {};
  for (int kt = 0; kt < nkt; kt++) {
    const int k0 = kt * 32;
    u16x16 au0, au1;
    {
      u16x8 lo = *(const u16x8*)(arow0 + k0 + ka);
      u16x8 hi = *(const u16x8*)(arow0 + k0 + ka + 16);
#pragma unroll
      for (int i = 0; i < 8; i++) { au0[i] = lo[i]; au0[i + 8] = hi[i]; }
    }
    {
      u16x8 lo = *(const u16x8*)(arow1 + k0 + ka);
      u16x8 hi = *(const u16x8*)(arow1 + k0 + ka + 16);
#pragma unroll
      for (int i = 0; i < 8; i++) { au1[i] = lo[i]; au1[i + 8] = hi[i]; }
    }
    v16bf a0 = __builtin_bit_cast(v16bf, au0);
    v16bf a1 = __builtin_bit_cast(v16bf, au1);

#pragma unroll
    for (int t = 0; t < 4; t++) {
      const unsigned short* bp = BP + ((size_t)(kt * 16 + colt0 + t) * 32 + lane) * 16;
      v16bf b = __builtin_bit_cast(v16bf, *(const u16x16*)bp);
      acc[0][t] = __builtin_amdgcn_wmma_f32_16x16x32_bf16(false, a0, false, b,
                                                          (short)0, acc[0][t], false, false);
      acc[1][t] = __builtin_amdgcn_wmma_f32_16x16x32_bf16(false, a1, false, b,
                                                          (short)0, acc[1][t], false, false);
    }
  }

  const int mlo = (lane < 16) ? 0 : 8;
#pragma unroll
  for (int mt = 0; mt < 2; mt++) {
    const int mbase = row0 + mt * 16 + mlo;
#pragma unroll
    for (int t = 0; t < 4; t++)
#pragma unroll
      for (int r = 0; r < 8; r++) {
        int row = mbase + r;
        if (row < NN) C[(size_t)row * 256 + (colt0 + t) * 16 + n] = acc[mt][t][r];
      }
  }
}

// ---------------- attention logits + softmax-state init ----------------
__global__ void k_attn_logits(const float* __restrict__ FT, const float* __restrict__ al,
                              const float* __restrict__ ar, float* __restrict__ EL,
                              float* __restrict__ ER, unsigned* __restrict__ MK,
                              float* __restrict__ DEN) {
  int i = blockIdx.x * blockDim.x + threadIdx.x;
  if (i >= NN * NH) return;
  int nidx = i >> 2, h = i & 3;
  const float* f = FT + (size_t)nidx * HDIM + h * HIDD;
  const float* a = al + h * HIDD;
  const float* b = ar + h * HIDD;
  float sl = 0.f, sr = 0.f;
  for (int d = 0; d < HIDD; d++) { float v = f[d]; sl += v * a[d]; sr += v * b[d]; }
  EL[i] = sl; ER[i] = sr; MK[i] = 0u; DEN[i] = 0.f;
}

__global__ void k_edge_max(const int* __restrict__ src, const int* __restrict__ dst,
                           const float* __restrict__ EL, const float* __restrict__ ER,
                           float* __restrict__ EB, unsigned* __restrict__ MK) {
  int i = blockIdx.x * blockDim.x + threadIdx.x;
  if (i >= NE * NH) return;
  int e = i >> 2, h = i & 3;
  float v = EL[src[e] * NH + h] + ER[dst[e] * NH + h];
  v = v > 0.f ? v : 0.2f * v;                   // leaky_relu(0.2)
  EB[i] = v;
  atomicMax(&MK[dst[e] * NH + h], fenc(v));
}

__global__ void k_edge_expsum(const int* __restrict__ dst, const unsigned* __restrict__ MK,
                              float* __restrict__ EB, float* __restrict__ DEN) {
  int i = blockIdx.x * blockDim.x + threadIdx.x;
  if (i >= NE * NH) return;
  int e = i >> 2, h = i & 3;
  float ex = __expf(EB[i] - fdec(MK[dst[e] * NH + h]));
  EB[i] = ex;
  atomicAdd(&DEN[dst[e] * NH + h], ex);
}

__global__ void k_edge_scatter(const int* __restrict__ src, const int* __restrict__ dst,
                               const float* __restrict__ EB, const float* __restrict__ DEN,
                               const float* __restrict__ FT, float* __restrict__ OUT) {
  int i = blockIdx.x * blockDim.x + threadIdx.x;
  if (i >= NE * HIDD) return;
  int e = i >> 6, d = i & 63;
  int s = src[e], t = dst[e];
#pragma unroll
  for (int h = 0; h < NH; h++) {
    float alpha = EB[e * NH + h] / DEN[t * NH + h];
    atomicAdd(&OUT[(size_t)t * HDIM + h * HIDD + d],
              alpha * FT[(size_t)s * HDIM + h * HIDD + d]);
  }
}

__global__ void k_headmean(const float* __restrict__ X, float* __restrict__ HF) {
  int i = blockIdx.x * blockDim.x + threadIdx.x;
  if (i >= NN * HIDD) return;
  int nidx = i >> 6, d = i & 63;
  const float* p = X + (size_t)nidx * HDIM + d;
  HF[i] = 0.25f * (p[0] + p[64] + p[128] + p[192]);
}

// ---------------- fused WMMA edge scorer ----------------
// wave handles 16 edges: A = |hf[src]-hf[dst]| (16x64 bf16, built in regs),
// B = Wm1 packed bf16 (64x64), z = relu(A@B + bm1), score = sigmoid(z@Wm2 + bm2).
__global__ void k_edge_score(const float* __restrict__ HF, const int* __restrict__ src,
                             const int* __restrict__ dst, const unsigned short* __restrict__ WBP,
                             const float* __restrict__ bm1, const float* __restrict__ Wm2,
                             const float* __restrict__ bm2, float* __restrict__ score) {
  __shared__ float zsh[4][16][64];
  const int lane = threadIdx.x & 31;
  const int wave = threadIdx.x >> 5;
  const int row  = lane & 15;
  const int ka   = (lane >= 16) ? 8 : 0;
  const int e    = (blockIdx.x * 4 + wave) * 16 + row;  // NE % 64 == 0
  const int s = src[e], t = dst[e];
  const float* ps = HF + (size_t)s * HIDD;
  const float* pt = HF + (size_t)t * HIDD;

  v8f acc[4] = {};
#pragma unroll
  for (int kt = 0; kt < 2; kt++) {
    const int k0 = kt * 32;
    u16x16 au;
#pragma unroll
    for (int i = 0; i < 8; i++) {
      int k = k0 + ka + i;
      au[i]     = f2bf(fabsf(ps[k]      - pt[k]));
      au[i + 8] = f2bf(fabsf(ps[k + 16] - pt[k + 16]));
    }
    v16bf a = __builtin_bit_cast(v16bf, au);
#pragma unroll
    for (int tl = 0; tl < 4; tl++) {
      const unsigned short* bp = WBP + ((size_t)(kt * 4 + tl) * 32 + lane) * 16;
      v16bf b = __builtin_bit_cast(v16bf, *(const u16x16*)bp);
      acc[tl] = __builtin_amdgcn_wmma_f32_16x16x32_bf16(false, a, false, b,
                                                        (short)0, acc[tl], false, false);
    }
  }
  const int moff = (lane < 16) ? 0 : 8;
#pragma unroll
  for (int tl = 0; tl < 4; tl++)
#pragma unroll
    for (int r = 0; r < 8; r++) {
      int col = 16 * tl + row;
      float z = acc[tl][r] + bm1[col];
      zsh[wave][moff + r][col] = z > 0.f ? z : 0.f;
    }
  __syncthreads();
  // split the 64-dot across the two half-waves, combine with wave32 shuffle
  float sum = 0.f;
  const int hoff = (lane >> 4) * 32;
#pragma unroll
  for (int d = 0; d < 32; d++) sum += zsh[wave][row][hoff + d] * Wm2[hoff + d];
  sum += __shfl_xor(sum, 16, 32);
  if (lane < 16) score[e] = 1.f / (1.f + __expf(-(sum + bm2[0])));
}

// ---------------- orchestration ----------------
extern "C" void kernel_launch(void* const* d_in, const int* in_sizes, int n_in,
                              void* d_out, int out_size, void* d_ws, size_t ws_size,
                              hipStream_t stream) {
  (void)in_sizes; (void)n_in; (void)out_size; (void)ws_size;

  const float* h   = (const float*)d_in[0];
  const int*   src = (const int*)  d_in[1];
  const int*   dst = (const int*)  d_in[2];
  const float* W1  = (const float*)d_in[3];
  const float* Wr1 = (const float*)d_in[4];
  const float* al1 = (const float*)d_in[5];
  const float* ar1 = (const float*)d_in[6];
  const float* W2  = (const float*)d_in[7];
  const float* al2 = (const float*)d_in[8];
  const float* ar2 = (const float*)d_in[9];
  const float* W3  = (const float*)d_in[10];
  const float* al3 = (const float*)d_in[11];
  const float* ar3 = (const float*)d_in[12];
  const float* Wm1 = (const float*)d_in[13];
  const float* bm1 = (const float*)d_in[14];
  const float* Wm2 = (const float*)d_in[15];
  const float* bm2 = (const float*)d_in[16];
  float* out = (float*)d_out;

  // workspace layout (all offsets multiples of 256 B; total ~208 MB)
  char* ws = (char*)d_ws;
  size_t off = 0;
  auto alloc = [&](size_t bytes) { void* p = ws + off; off += (bytes + 255) & ~(size_t)255; return p; };
  unsigned short* XB  = (unsigned short*)alloc((size_t)NN * HDIM * 2);   // bf16 activations
  float*          FT  = (float*)alloc((size_t)NN * HDIM * 4);
  float*          P0  = (float*)alloc((size_t)NN * HDIM * 4);
  float*          P1  = (float*)alloc((size_t)NN * HDIM * 4);
  unsigned short* WBa = (unsigned short*)alloc((size_t)HDIM * HDIM * 2); // packed weights
  unsigned short* WBb = (unsigned short*)alloc((size_t)HDIM * HDIM * 2);
  unsigned short* WBm = (unsigned short*)alloc((size_t)HIDD * HIDD * 2);
  float*          EL  = (float*)alloc((size_t)NN * NH * 4);
  float*          ER  = (float*)alloc((size_t)NN * NH * 4);
  unsigned*       MK  = (unsigned*)alloc((size_t)NN * NH * 4);
  float*          DEN = (float*)alloc((size_t)NN * NH * 4);
  float*          EB  = (float*)alloc((size_t)NE * NH * 4);
  float*          HF  = (float*)alloc((size_t)NN * HIDD * 4);

  auto blks = [](long n, int b) { return (int)((n + b - 1) / b); };
  const int GEMM_GRID = (NN + 31) / 32;  // 1563 (last block row-guarded on store)

  auto gat_edges = [&](const float* alp, const float* arp, float* OUT) {
    k_attn_logits<<<blks((long)NN * NH, 256), 256, 0, stream>>>(FT, alp, arp, EL, ER, MK, DEN);
    k_edge_max   <<<blks((long)NE * NH, 256), 256, 0, stream>>>(src, dst, EL, ER, EB, MK);
    k_edge_expsum<<<blks((long)NE * NH, 256), 256, 0, stream>>>(dst, MK, EB, DEN);
    k_edge_scatter<<<blks((long)NE * HIDD, 256), 256, 0, stream>>>(src, dst, EB, DEN, FT, OUT);
  };

  // ----- layer 1: ft = h@W1, res = h@Wr1 -----
  k_cvt_bf16<<<blks((long)NN * INF_, 256), 256, 0, stream>>>(h, XB, NN * INF_);
  k_pack_w<<<blks((INF_ / 32) * 16 * 32, 256), 256, 0, stream>>>(W1,  WBa, INF_, HDIM);
  k_pack_w<<<blks((INF_ / 32) * 16 * 32, 256), 256, 0, stream>>>(Wr1, WBb, INF_, HDIM);
  k_gemm_bf16_n256<<<GEMM_GRID, 128, 0, stream>>>(XB, WBa, FT, INF_);
  k_gemm_bf16_n256<<<GEMM_GRID, 128, 0, stream>>>(XB, WBb, P0, INF_);  // residual preloaded
  gat_edges(al1, ar1, P0);
  k_relu_cvt<<<blks((long)NN * HDIM, 256), 256, 0, stream>>>(P0, XB, NN * HDIM);

  // ----- layer 2: identity residual -----
  k_pack_w<<<blks((HDIM / 32) * 16 * 32, 256), 256, 0, stream>>>(W2, WBa, HDIM, HDIM);
  k_gemm_bf16_n256<<<GEMM_GRID, 128, 0, stream>>>(XB, WBa, FT, HDIM);
  hipMemcpyAsync(P1, P0, (size_t)NN * HDIM * 4, hipMemcpyDeviceToDevice, stream);
  gat_edges(al2, ar2, P1);
  k_relu_cvt<<<blks((long)NN * HDIM, 256), 256, 0, stream>>>(P1, XB, NN * HDIM);

  // ----- layer 3: identity residual, no activation -----
  k_pack_w<<<blks((HDIM / 32) * 16 * 32, 256), 256, 0, stream>>>(W3, WBa, HDIM, HDIM);
  k_gemm_bf16_n256<<<GEMM_GRID, 128, 0, stream>>>(XB, WBa, FT, HDIM);
  hipMemcpyAsync(P0, P1, (size_t)NN * HDIM * 4, hipMemcpyDeviceToDevice, stream);
  gat_edges(al3, ar3, P0);

  // ----- head mean + fused WMMA edge scorer -----
  k_headmean<<<blks((long)NN * HIDD, 256), 256, 0, stream>>>(P0, HF);
  k_pack_w<<<blks((HIDD / 32) * (HIDD / 16) * 32, 256), 256, 0, stream>>>(Wm1, WBm, HIDD, HIDD);
  k_edge_score<<<NE / 64, 128, 0, stream>>>(HF, src, dst, WBm, bm1, Wm2, bm2, out);
}
